// TransformerLayerController_55508157333915
// MI455X (gfx1250) — compile-verified, matrix operand order checked by reading, not run on hardware
//
#include <hip/hip_runtime.h>
#include <math.h>

// ---------------------------------------------------------------------------
// Problem constants (from reference)
// ---------------------------------------------------------------------------
constexpr int B_ = 2, H_ = 32, S_ = 2048, D_ = 128;
constexpr int SINK = 4, K_OUT = 16, V_OUT = 4;
constexpr float QMAX = 127.0f;

constexpr size_t NE = (size_t)B_ * H_ * S_ * D_;       // 16,777,216
// output offsets (floats) in d_out, reference return order
constexpr size_t OFF_ATT  = 0;
constexpr size_t OFF_KQ   = NE;
constexpr size_t OFF_KS   = 2 * NE;                    // k_scale [B,H,1,D]
constexpr size_t OFF_VQ   = OFF_KS  + (size_t)B_ * H_ * D_;
constexpr size_t OFF_VS   = OFF_VQ  + NE;              // v_scale [B,H,S,1]
constexpr size_t OFF_KSPV = OFF_VS  + (size_t)B_ * H_ * S_;
constexpr size_t OFF_KSPI = OFF_KSPV + (size_t)B_ * H_ * D_ * K_OUT;
constexpr size_t OFF_VSPV = OFF_KSPI + (size_t)B_ * H_ * D_ * K_OUT;
constexpr size_t OFF_VSPI = OFF_VSPV + (size_t)B_ * H_ * S_ * V_OUT;

// ---------------------------------------------------------------------------
// WMMA types / helpers (CDNA5 gfx1250, wave32)
// ---------------------------------------------------------------------------
typedef __attribute__((ext_vector_type(16))) _Float16 v16h;
typedef __attribute__((ext_vector_type(8)))  _Float16 v8h;
typedef __attribute__((ext_vector_type(8)))  float    v8f;

__device__ __forceinline__ v8f wmma32(v16h a, v16h b, v8f c) {
  // D(16x16 f32) = A(16x32 f16) * B(32x16 f16) + C
  return __builtin_amdgcn_wmma_f32_16x16x32_f16(false, a, false, b,
                                                (short)0, c, false, false);
}

// two contiguous 16B chunks -> one A/B-operand v16h (ISA 16-bit 16x32 layout)
__device__ __forceinline__ v16h lds_frag(const _Float16* p) {
  v8h lo = *reinterpret_cast<const v8h*>(p);
  v8h hi = *reinterpret_cast<const v8h*>(p + 16);
  return __builtin_shufflevector(lo, hi, 0, 1, 2, 3, 4, 5, 6, 7,
                                 8, 9, 10, 11, 12, 13, 14, 15);
}

// RoPE cos/sin at position p, dim d (base 10000, D=128 -> half=64)
__device__ __forceinline__ void rope_cs(int p, int d, float& c, float& s) {
  // inv_freq = 10000^(-(d%64)/64) = exp(-(d%64) * ln(1e4)/64)
  float inv = __expf(-(float)(d & 63) * 0.14391156831212787f);
  float th  = (float)p * inv;
  c = __cosf(th);
  s = __sinf(th);
}

// ---------------------------------------------------------------------------
// Kernel 1: flash attention with RoPE-on-load, WMMA f16->f32
//   grid = (S/128, B*H), block = 256 (8 waves x wave32), each wave: 16 q rows
//   BM=128 halves K/V re-read traffic vs BM=64; K/V tiles staged once in LDS
//   and consumed by all 8 waves.
// ---------------------------------------------------------------------------
constexpr int BcT  = 32;    // keys per tile
constexpr int BMQ  = 128;   // query rows per block
constexpr int NWAV = 8;
constexpr int KSTR = 136;   // LDS row strides in halfs (16B aligned, bank-spread)
constexpr int VSTR = 40;
constexpr int PSTR = 40;

__global__ __launch_bounds__(256)
void fa_kernel(const float* __restrict__ q, const float* __restrict__ k,
               const float* __restrict__ v, const int* __restrict__ pos,
               float* __restrict__ out) {
  __shared__ __align__(16) _Float16 kt[BcT * KSTR];        // K tile, row-major
  __shared__ __align__(16) _Float16 vt[D_ * VSTR];         // V tile, transposed
  __shared__ __align__(16) _Float16 plds[NWAV * 16 * PSTR];// per-wave P buffer

  const int bh   = blockIdx.y;
  const int b    = bh >> 5;
  const int q0   = blockIdx.x * BMQ;
  const int tid  = threadIdx.x;
  const int wave = tid >> 5;
  const int lane = tid & 31;
  const int half = lane >> 4;
  const int lrow = lane & 15;

  const size_t base = (size_t)bh * S_ * D_;
  const float* qb = q + base;
  const float* kb = k + base;
  const float* vb = v + base;

  // ---- load this wave's 16 Q rows (RoPE + 1/sqrt(D)) in A-operand layout ----
  const int   qrow = q0 + wave * 16 + lrow;
  const float* qr  = qb + (size_t)qrow * D_;
  const int   qp   = pos[b * S_ + qrow];
  v16h afrag[4];
#pragma unroll
  for (int t = 0; t < 4; ++t) {
#pragma unroll
    for (int j = 0; j < 16; ++j) {
      int dd = t * 32 + ((j < 8) ? 0 : 16) + half * 8 + (j & 7);
      float cc, ss;
      rope_cs(qp, dd, cc, ss);
      float x  = qr[dd];
      float xr = (dd < 64) ? -qr[dd + 64] : qr[dd - 64];
      afrag[t][j] = (_Float16)((x * cc + xr * ss) * 0.08838834764831845f);
    }
  }

  const v8f vzero = {0.f, 0.f, 0.f, 0.f, 0.f, 0.f, 0.f, 0.f};
  v8f acc[8];
#pragma unroll
  for (int f = 0; f < 8; ++f) acc[f] = vzero;
  float m_i[8], l_i[8];
#pragma unroll
  for (int r = 0; r < 8; ++r) { m_i[r] = -1e30f; l_i[r] = 0.f; }

  const int ntiles = (q0 + BMQ) / BcT;   // causal upper bound
  for (int t = 0; t < ntiles; ++t) {
    const int k0 = t * BcT;
    __syncthreads();
    // ---- stage K (RoPE'd, row-major) and V (transposed) tiles as f16 ----
    for (int i = tid; i < BcT * D_; i += 256) {
      int kk = i >> 7, dd = i & 127;
      int srow = k0 + kk;
      const float* kr = kb + (size_t)srow * D_;
      float cc, ss;
      rope_cs(pos[b * S_ + srow], dd, cc, ss);
      float x  = kr[dd];
      float xr = (dd < 64) ? -kr[dd + 64] : kr[dd - 64];
      kt[kk * KSTR + dd] = (_Float16)(x * cc + xr * ss);
      vt[dd * VSTR + kk] = (_Float16)(vb[(size_t)srow * D_ + dd]);
      // pull next tile toward the near cache while this tile's WMMAs run
      if (t + 1 < ntiles) {
        __builtin_prefetch(kb + (size_t)(srow + BcT) * D_ + dd, 0, 1);
        __builtin_prefetch(vb + (size_t)(srow + BcT) * D_ + dd, 0, 1);
      }
    }
    __syncthreads();

    // ---- scores: 16 q-rows x 32 keys (two 16x16 frags), K-dim = 128 ----
    v8f c0 = vzero, c1 = vzero;
#pragma unroll
    for (int dt = 0; dt < 4; ++dt) {
      v16h bk0 = lds_frag(&kt[lrow * KSTR + dt * 32 + half * 8]);
      v16h bk1 = lds_frag(&kt[(16 + lrow) * KSTR + dt * 32 + half * 8]);
      c0 = wmma32(afrag[dt], bk0, c0);
      c1 = wmma32(afrag[dt], bk1, c1);
    }

    // ---- causal mask (wave-uniform branch; per-element VALU select) ----
    if (k0 + BcT - 1 > q0 + wave * 16) {
#pragma unroll
      for (int r = 0; r < 8; ++r) {
        int qi = q0 + wave * 16 + r + 8 * half;
        if (k0 + lrow      > qi) c0[r] = -1e30f;
        if (k0 + 16 + lrow > qi) c1[r] = -1e30f;
      }
    }

    // ---- online softmax (row = r + 8*half, cols across 16 lanes of half) ----
    _Float16* pw = &plds[wave * 16 * PSTR];
#pragma unroll
    for (int r = 0; r < 8; ++r) {
      float mx = fmaxf(c0[r], c1[r]);
#pragma unroll
      for (int off = 8; off >= 1; off >>= 1) mx = fmaxf(mx, __shfl_xor(mx, off));
      float mn   = fmaxf(m_i[r], mx);
      float corr = __expf(m_i[r] - mn);
      float p0   = __expf(c0[r] - mn);
      float p1   = __expf(c1[r] - mn);
      float ps   = p0 + p1;
#pragma unroll
      for (int off = 8; off >= 1; off >>= 1) ps += __shfl_xor(ps, off);
      l_i[r] = l_i[r] * corr + ps;
      m_i[r] = mn;
#pragma unroll
      for (int f = 0; f < 8; ++f) acc[f][r] = acc[f][r] * corr;
      pw[(r + 8 * half) * PSTR + lrow]      = (_Float16)p0;
      pw[(r + 8 * half) * PSTR + 16 + lrow] = (_Float16)p1;
    }

    // ---- PV: P(16x32) x V(32x128), 8 WMMAs accumulate into O ----
    v16h pa = lds_frag(&pw[lrow * PSTR + half * 8]);
#pragma unroll
    for (int f = 0; f < 8; ++f) {
      v16h bv = lds_frag(&vt[(f * 16 + lrow) * VSTR + half * 8]);
      acc[f] = wmma32(pa, bv, acc[f]);
    }
  }

  // ---- epilogue: O / l ----
#pragma unroll
  for (int f = 0; f < 8; ++f) {
#pragma unroll
    for (int r = 0; r < 8; ++r) {
      int row = q0 + wave * 16 + r + 8 * half;
      int dd  = f * 16 + lrow;
      out[OFF_ATT + base + (size_t)row * D_ + dd] = acc[f][r] / l_i[r];
    }
  }
}

// ---------------------------------------------------------------------------
// Kernel 2: K outlier isolation + per-channel int8 quant
//   one 256-thread block per (b,h,d) channel; seq length 2048 -> 8 regs/thread
// ---------------------------------------------------------------------------
__global__ __launch_bounds__(256)
void kout_kernel(const float* __restrict__ k, const int* __restrict__ pos,
                 float* __restrict__ out) {
  const int ch = blockIdx.x;          // ch = bh*128 + d
  const int d  = ch & 127;
  const int bh = ch >> 7;
  const int b  = ch >> 12;
  const int tid = threadIdx.x;
  const float* kb = k + (size_t)bh * S_ * D_;

  float val[8], av[8];
#pragma unroll
  for (int i = 0; i < 8; ++i) {
    int s = tid + i * 256;
    float cc, ss;
    rope_cs(pos[b * S_ + s], d, cc, ss);
    float x  = kb[(size_t)s * D_ + d];
    float xr = (d < 64) ? -kb[(size_t)s * D_ + d + 64] : kb[(size_t)s * D_ + d - 64];
    float r  = x * cc + xr * ss;
    if (s < SINK) r = 0.f;            // sink tokens excluded / zeroed
    val[i] = r;
    av[i]  = fabsf(r);
  }

  __shared__ float sa[256];
  __shared__ float sv[256];
  __shared__ int   si[256];

  for (int o = 0; o < K_OUT; ++o) {
    // local argmax (strict > keeps lowest index on ties, idx grows with i)
    float bm = av[0]; float bs = val[0]; int bi = tid;
#pragma unroll
    for (int i = 1; i < 8; ++i)
      if (av[i] > bm) { bm = av[i]; bs = val[i]; bi = tid + i * 256; }
    sa[tid] = bm; sv[tid] = bs; si[tid] = bi;
    __syncthreads();
    for (int st = 128; st > 0; st >>= 1) {
      if (tid < st) {
        float oa = sa[tid + st]; int oi = si[tid + st];
        if (oa > sa[tid] || (oa == sa[tid] && oi < si[tid])) {
          sa[tid] = oa; si[tid] = oi; sv[tid] = sv[tid + st];
        }
      }
      __syncthreads();
    }
    int   widx = si[0];
    float wval = sv[0];
    if (tid == 0) {
      out[OFF_KSPV + (size_t)ch * K_OUT + o] = wval;
      out[OFF_KSPI + (size_t)ch * K_OUT + o] = (float)widx;
    }
    if ((widx & 255) == tid) {        // owner removes it from dense + candidates
      int i = widx >> 8;
      val[i] = 0.f; av[i] = -1.f;
    }
    __syncthreads();
  }

  // per-channel absmax over dense (outliers + sink already zeroed)
  float mx = 0.f;
#pragma unroll
  for (int i = 0; i < 8; ++i) mx = fmaxf(mx, fabsf(val[i]));
  sa[tid] = mx;
  __syncthreads();
  for (int st = 128; st > 0; st >>= 1) {
    if (tid < st) sa[tid] = fmaxf(sa[tid], sa[tid + st]);
    __syncthreads();
  }
  float scale = fmaxf(sa[0], 1e-8f) * (1.f / QMAX);
  if (tid == 0) out[OFF_KS + ch] = scale;   // [B,H,1,D] flat == ch

#pragma unroll
  for (int i = 0; i < 8; ++i) {
    int s = tid + i * 256;
    float qv = fminf(fmaxf(rintf(val[i] / scale), -QMAX), QMAX);
    out[OFF_KQ + (size_t)bh * S_ * D_ + (size_t)s * D_ + d] = qv;
  }
}

// ---------------------------------------------------------------------------
// Kernel 3: V outlier isolation + per-token int8 quant
//   one wave32 per token (b,h,s); D=128 -> 4 regs/lane
// ---------------------------------------------------------------------------
__global__ __launch_bounds__(256)
void vout_kernel(const float* __restrict__ v, float* __restrict__ out) {
  const int token = blockIdx.x * 8 + (threadIdx.x >> 5); // flat (b,h,s)
  const int lane  = threadIdx.x & 31;
  const int s     = token & (S_ - 1);
  const float* vr = v + (size_t)token * D_;

  float val[4], av[4];
#pragma unroll
  for (int i = 0; i < 4; ++i) {
    int dd = lane + 32 * i;
    float x = (s < SINK) ? 0.f : vr[dd];
    val[i] = x;
    av[i]  = fabsf(x);
  }

  for (int o = 0; o < V_OUT; ++o) {
    float bm = av[0]; float bs = val[0]; int bi = lane;
#pragma unroll
    for (int i = 1; i < 4; ++i)
      if (av[i] > bm) { bm = av[i]; bs = val[i]; bi = lane + 32 * i; }
#pragma unroll
    for (int off = 16; off >= 1; off >>= 1) {
      float om = __shfl_xor(bm, off);
      float os = __shfl_xor(bs, off);
      int   oi = __shfl_xor(bi, off);
      if (om > bm || (om == bm && oi < bi)) { bm = om; bs = os; bi = oi; }
    }
    if (lane == 0) {
      out[OFF_VSPV + (size_t)token * V_OUT + o] = bs;
      out[OFF_VSPI + (size_t)token * V_OUT + o] = (float)bi;
    }
    if ((bi & 31) == lane) {          // owner removes winner
      int i = bi >> 5;
      val[i] = 0.f; av[i] = -1.f;
    }
  }

  float mx = 0.f;
#pragma unroll
  for (int i = 0; i < 4; ++i) mx = fmaxf(mx, fabsf(val[i]));
#pragma unroll
  for (int off = 16; off >= 1; off >>= 1) mx = fmaxf(mx, __shfl_xor(mx, off));
  float scale = fmaxf(mx, 1e-8f) * (1.f / QMAX);
  if (lane == 0) out[OFF_VS + token] = scale;

#pragma unroll
  for (int i = 0; i < 4; ++i) {
    int dd = lane + 32 * i;
    float qv = fminf(fmaxf(rintf(val[i] / scale), -QMAX), QMAX);
    out[OFF_VQ + (size_t)token * D_ + dd] = qv;
  }
}

// ---------------------------------------------------------------------------
// Launcher
// ---------------------------------------------------------------------------
extern "C" void kernel_launch(void* const* d_in, const int* in_sizes, int n_in,
                              void* d_out, int out_size, void* d_ws, size_t ws_size,
                              hipStream_t stream) {
  (void)in_sizes; (void)n_in; (void)out_size; (void)d_ws; (void)ws_size;
  const float* q   = (const float*)d_in[0];
  const float* k   = (const float*)d_in[1];
  const float* v   = (const float*)d_in[2];
  const int*   pos = (const int*)d_in[3];
  float* out = (float*)d_out;

  dim3 fg(S_ / BMQ, B_ * H_);                         // 16 x 64 blocks
  fa_kernel<<<fg, 256, 0, stream>>>(q, k, v, pos, out);
  kout_kernel<<<B_ * H_ * D_, 256, 0, stream>>>(k, pos, out);   // 8192 blocks
  vout_kernel<<<(B_ * H_ * S_) / 8, 256, 0, stream>>>(v, out);  // 16384 blocks
}
